// DecoderAttention_70222715289878
// MI455X (gfx1250) — matrix operand order repaired and checked by reading, not compiled, not run on hardware
//
#include <hip/hip_runtime.h>

#define B_  32
#define K_  8
#define LQ_ 128
#define LR_ 256
#define H_  512
#define L_  2176
#define BK_ 256

typedef __attribute__((ext_vector_type(16))) __bf16 v16bf;
typedef __attribute__((ext_vector_type(8)))  float  v8f;

union AB { v16bf v; uint4 u[2]; };

__device__ __forceinline__ unsigned short f32_to_bf16(float f) {
  union { float f; unsigned u; } a; a.f = f;
  return (unsigned short)((a.u + 0x7FFFu + ((a.u >> 16) & 1u)) >> 16);
}
// monotone float -> u32 encoding so max works with integer atomics (deterministic)
__device__ __forceinline__ unsigned fenc(float f) {
  union { float f; unsigned u; } a; a.f = f;
  return (a.u & 0x80000000u) ? ~a.u : (a.u | 0x80000000u);
}
__device__ __forceinline__ float fdec(unsigned e) {
  union { float f; unsigned u; } a;
  a.u = (e & 0x80000000u) ? (e & 0x7FFFFFFFu) : ~e;
  return a.f;
}

__global__ void k_convert_bf16(const float* __restrict__ src,
                               unsigned short* __restrict__ dst, int n) {
  int i = blockIdx.x * blockDim.x + threadIdx.x;
  int stride = gridDim.x * blockDim.x;
  for (; i < n; i += stride) dst[i] = f32_to_bf16(src[i]);
}

// ---- K1: per (b,k): hqU = hq@Uw^T (LDS), U = tanh(hqU@hr^T) row/col max, softmaxes
__global__ __launch_bounds__(256) void k1_coatt(
    const unsigned short* __restrict__ hq_bf,
    const unsigned short* __restrict__ hr_bf,
    const unsigned short* __restrict__ Uw_bf,
    float* __restrict__ alpha_q, float* __restrict__ alpha_r) {
  extern __shared__ char smem[];
  unsigned short* hqU  = (unsigned short*)smem;                       // 128*520 bf16
  unsigned* rowmax     = (unsigned*)(smem + 128 * 520 * 2);           // 128
  unsigned* colmax     = (unsigned*)(smem + 128 * 520 * 2 + 512);     // 256
  float* red           = (float*)(smem + 128 * 520 * 2 + 512 + 1024); // 256

  const int tid = threadIdx.x;
  const int lane = tid & 31, w = tid >> 5;
  const int ln = lane & 15, hi = lane >> 4;
  const int bk = blockIdx.x, b = bk >> 3;

  const unsigned short* hq = hq_bf + (size_t)b * LQ_ * H_;
  const unsigned short* hr = hr_bf + (size_t)bk * LR_ * H_;

  if (tid < 128) rowmax[tid] = 0u;
  colmax[tid] = 0u;

  const v8f vzero = {0.f,0.f,0.f,0.f,0.f,0.f,0.f,0.f};
  const int koff  = hi ? 8 : 0;   // A-operand: interleaved 8+8 K runs
  const int kbase = hi ? 8 : 0;   // B-operand: interleaved 8+8 K runs
  const int rowA = w * 16 + ln;

  // Phase 1: hqU[16w..16w+15][0..511]
  for (int nc = 0; nc < 8; ++nc) {
    const int o0 = nc * 64;
    v8f acc[4]; for (int t = 0; t < 4; ++t) acc[t] = vzero;
    for (int kb = 0; kb < 16; ++kb) {
      const int i0 = kb * 32;
      AB A;
      const unsigned short* pa = hq + (size_t)rowA * H_ + i0 + koff;
      A.u[0] = *(const uint4*)pa;
      A.u[1] = *(const uint4*)(pa + 16);
      for (int t = 0; t < 4; ++t) {
        AB Bm;
        const unsigned short* pb = Uw_bf + (size_t)(o0 + t * 16 + ln) * H_ + i0 + kbase;
        Bm.u[0] = *(const uint4*)pb;
        Bm.u[1] = *(const uint4*)(pb + 16);
        acc[t] = __builtin_amdgcn_wmma_f32_16x16x32_bf16(
            false, A.v, false, Bm.v, (short)0, acc[t], false, false);
      }
    }
    for (int t = 0; t < 4; ++t)
      for (int v = 0; v < 8; ++v)
        hqU[(size_t)(w * 16 + v + 8 * hi) * 520 + o0 + t * 16 + ln] =
            f32_to_bf16(acc[t][v]);
  }
  __syncthreads();

  // Phase 2: U = tanh(hqU @ hr^T) with running row/col max
  for (int rc = 0; rc < 4; ++rc) {
    const int r0 = rc * 64;
    v8f acc[4]; for (int t = 0; t < 4; ++t) acc[t] = vzero;
    for (int kb = 0; kb < 16; ++kb) {
      const int o0 = kb * 32;
      AB A;
      const unsigned short* pa = hqU + (size_t)rowA * 520 + o0 + koff;
      A.u[0] = *(const uint4*)pa;
      A.u[1] = *(const uint4*)(pa + 16);
      for (int t = 0; t < 4; ++t) {
        AB Bm;
        const unsigned short* pb = hr + (size_t)(r0 + t * 16 + ln) * H_ + o0 + kbase;
        Bm.u[0] = *(const uint4*)pb;
        Bm.u[1] = *(const uint4*)(pb + 16);
        acc[t] = __builtin_amdgcn_wmma_f32_16x16x32_bf16(
            false, A.v, false, Bm.v, (short)0, acc[t], false, false);
      }
    }
    for (int t = 0; t < 4; ++t) {
      float cm = -2.f;
      for (int v = 0; v < 8; ++v) {
        float val = tanhf(acc[t][v]);
        atomicMax(&rowmax[w * 16 + v + 8 * hi], fenc(val));
        cm = fmaxf(cm, val);
      }
      atomicMax(&colmax[r0 + t * 16 + ln], fenc(cm));
    }
  }
  __syncthreads();

  // Phase 3: softmaxes (tanh-bounded -> no max subtraction needed)
  float eq = 0.f;
  if (tid < 128) eq = expf(fdec(rowmax[tid]));
  float er = expf(fdec(colmax[tid]));
  red[tid] = (tid < 128) ? eq : 0.f;
  __syncthreads();
  for (int s = 128; s > 0; s >>= 1) { if (tid < s) red[tid] += red[tid + s]; __syncthreads(); }
  float sq = red[0];
  __syncthreads();
  red[tid] = er;
  __syncthreads();
  for (int s = 128; s > 0; s >>= 1) { if (tid < s) red[tid] += red[tid + s]; __syncthreads(); }
  float sr = red[0];
  if (tid < 128) alpha_q[(size_t)bk * LQ_ + tid] = eq / sq;
  alpha_r[(size_t)bk * LR_ + tid] = er / sr;
}

// ---- K2a: per-row weights w[b,l]
__global__ void k2_wrow(const float* __restrict__ alpha_q, const float* __restrict__ alpha_r,
                        const float* __restrict__ q_mask, const float* __restrict__ r_mask,
                        float* __restrict__ wrow) {
  int idx = blockIdx.x * blockDim.x + threadIdx.x;
  if (idx >= B_ * L_) return;
  int b = idx / L_, l = idx % L_;
  float wv;
  if (l < LQ_) {
    float s = 0.f;
    for (int k = 0; k < K_; ++k) s += alpha_q[(size_t)(b * K_ + k) * LQ_ + l];
    wv = q_mask[b * LQ_ + l] * s * (1.f / K_);
  } else {
    int rr = l - LQ_;
    int k = rr >> 8, r = rr & 255;
    wv = r_mask[(size_t)b * K_ * LR_ + rr] * alpha_r[(size_t)(b * K_ + k) * LR_ + r];
  }
  wrow[idx] = wv;
}

// ---- K2b: sfea = s_t @ Ws^T (tiny, f32)
__global__ void k2_sfea(const float* __restrict__ s_t, const float* __restrict__ Ws,
                        float* __restrict__ sfea) {
  int idx = blockIdx.x * blockDim.x + threadIdx.x;
  if (idx >= B_ * H_) return;
  int b = idx / H_, o = idx % H_;
  const float4* a = (const float4*)(s_t + (size_t)b * H_);
  const float4* wr = (const float4*)(Ws + (size_t)o * H_);
  float acc = 0.f;
  for (int i = 0; i < H_ / 4; ++i) {
    float4 x = a[i], y = wr[i];
    acc += x.x * y.x + x.y * y.y + x.z * y.z + x.w * y.w;
  }
  sfea[idx] = acc;
}

// ---- K3: fused G = X @ Wqr^T, then logits = sum_o tanh(sfea + w*G + b + cov*Wc)*Vr
__global__ __launch_bounds__(256) void k3_logits(
    const unsigned short* __restrict__ hq_bf,
    const unsigned short* __restrict__ hr_bf,
    const unsigned short* __restrict__ Wqr_bf,
    const float* __restrict__ sfea, const float* __restrict__ Wqr_b,
    const float* __restrict__ Wc, const float* __restrict__ Vr,
    const float* __restrict__ wrow, const float* __restrict__ qr_cov,
    float* __restrict__ logits) {
  __shared__ float sfeaL[H_], biasL[H_], wcL[H_], vrL[H_];
  __shared__ float wrowL[16], covL[16];
  __shared__ float part[8][16];

  const int tid = threadIdx.x, lane = tid & 31, w = tid >> 5;
  const int ln = lane & 15, hi = lane >> 4;
  const int g = blockIdx.x;
  const int b = g / (L_ / 16);
  const int l0 = (g % (L_ / 16)) * 16;

  for (int i = tid; i < H_; i += 256) {
    sfeaL[i] = sfea[(size_t)b * H_ + i];
    biasL[i] = Wqr_b[i];
    wcL[i]   = Wc[i];
    vrL[i]   = Vr[i];
  }
  if (tid < 16) {
    wrowL[tid] = wrow[(size_t)b * L_ + l0 + tid];
    covL[tid]  = qr_cov[(size_t)b * L_ + l0 + tid];
  }
  __syncthreads();

  const unsigned short* asrc = (l0 < LQ_)
      ? hq_bf + ((size_t)b * LQ_ + l0) * H_
      : hr_bf + ((size_t)b * K_ * LR_ + (l0 - LQ_)) * H_;

  const int koff = hi ? 8 : 0, kbase = hi ? 8 : 0;
  const int o_base = w * 64;
  const v8f vzero = {0.f,0.f,0.f,0.f,0.f,0.f,0.f,0.f};
  v8f acc[4]; for (int t = 0; t < 4; ++t) acc[t] = vzero;

  for (int kb = 0; kb < 16; ++kb) {
    const int i0 = kb * 32;
    AB A;
    const unsigned short* pa = asrc + (size_t)ln * H_ + i0 + koff;
    A.u[0] = *(const uint4*)pa;
    A.u[1] = *(const uint4*)(pa + 16);
    for (int t = 0; t < 4; ++t) {
      AB Bm;
      const unsigned short* pb = Wqr_bf + (size_t)(o_base + t * 16 + ln) * H_ + i0 + kbase;
      Bm.u[0] = *(const uint4*)pb;
      Bm.u[1] = *(const uint4*)(pb + 16);
      acc[t] = __builtin_amdgcn_wmma_f32_16x16x32_bf16(
          false, A.v, false, Bm.v, (short)0, acc[t], false, false);
    }
  }

  for (int v = 0; v < 8; ++v) {
    const int row = v + 8 * hi;
    const float wr = wrowL[row], cv = covL[row];
    float s = 0.f;
    for (int t = 0; t < 4; ++t) {
      const int o = o_base + t * 16 + ln;
      float d = sfeaL[o] + wr * acc[t][v] + biasL[o] + cv * wcL[o];
      s += tanhf(d) * vrL[o];
    }
    for (int off = 1; off < 16; off <<= 1) s += __shfl_xor(s, off, 32);
    if (ln == 0) part[w][row] = s;
  }
  __syncthreads();
  if (tid < 16) {
    float s = 0.f;
    for (int ww = 0; ww < 8; ++ww) s += part[ww][tid];  // deterministic order
    logits[(size_t)b * L_ + l0 + tid] = s;
  }
}

// ---- K4: masked softmax over L, write a / new_cov, coef = a*w
__global__ __launch_bounds__(256) void k4_softmax(
    const float* __restrict__ logits, const float* __restrict__ q_mask,
    const float* __restrict__ r_mask, const float* __restrict__ qr_cov,
    const float* __restrict__ wrow, float* __restrict__ out_a,
    float* __restrict__ out_cov, float* __restrict__ coef) {
  __shared__ float red[256];
  const int tid = threadIdx.x, b = blockIdx.x;
  float x[9], e[9];
  float m = -1e30f;
  for (int j = 0; j < 9; ++j) {
    int idx = tid + 256 * j;
    x[j] = (idx < L_) ? logits[(size_t)b * L_ + idx] : -1e30f;
    m = fmaxf(m, x[j]);
  }
  red[tid] = m; __syncthreads();
  for (int s = 128; s > 0; s >>= 1) { if (tid < s) red[tid] = fmaxf(red[tid], red[tid + s]); __syncthreads(); }
  float gmax = red[0]; __syncthreads();
  float s1 = 0.f;
  for (int j = 0; j < 9; ++j) {
    int idx = tid + 256 * j;
    e[j] = (idx < L_) ? expf(x[j] - gmax) : 0.f;
    s1 += e[j];
  }
  red[tid] = s1; __syncthreads();
  for (int s = 128; s > 0; s >>= 1) { if (tid < s) red[tid] += red[tid + s]; __syncthreads(); }
  float sum1 = red[0]; __syncthreads();
  float s2 = 0.f;
  for (int j = 0; j < 9; ++j) {
    int idx = tid + 256 * j;
    if (idx < L_) {
      float mk = (idx < LQ_) ? q_mask[b * LQ_ + idx]
                             : r_mask[(size_t)b * K_ * LR_ + idx - LQ_];
      e[j] = e[j] / sum1 * mk;
      s2 += e[j];
    }
  }
  red[tid] = s2; __syncthreads();
  for (int s = 128; s > 0; s >>= 1) { if (tid < s) red[tid] += red[tid + s]; __syncthreads(); }
  float sum2 = red[0];
  for (int j = 0; j < 9; ++j) {
    int idx = tid + 256 * j;
    if (idx < L_) {
      float a = e[j] / sum2;
      out_a[(size_t)b * L_ + idx]  = a;
      out_cov[(size_t)b * L_ + idx] = qr_cov[(size_t)b * L_ + idx] + a;
      coef[(size_t)b * L_ + idx]   = a * wrow[(size_t)b * L_ + idx];
    }
  }
}

// ---- K5: c_t = coef . X  (f32 originals for precision)
__global__ __launch_bounds__(256) void k5_ct(
    const float* __restrict__ hq_f, const float* __restrict__ hr_f,
    const float* __restrict__ coef, float* __restrict__ c_t) {
  __shared__ float cL[L_];
  const int tid = threadIdx.x, b = blockIdx.x;
  for (int i = tid; i < L_; i += 256) cL[i] = coef[(size_t)b * L_ + i];
  __syncthreads();
  float a0 = 0.f, a1 = 0.f;
  const int h0 = tid, h1 = tid + 256;
  const float* hq = hq_f + (size_t)b * LQ_ * H_;
  const float* hr = hr_f + (size_t)b * K_ * LR_ * H_;
  for (int l = 0; l < LQ_; ++l) {
    float c = cL[l];
    a0 += c * hq[(size_t)l * H_ + h0];
    a1 += c * hq[(size_t)l * H_ + h1];
  }
  for (int r = 0; r < K_ * LR_; ++r) {
    float c = cL[LQ_ + r];
    a0 += c * hr[(size_t)r * H_ + h0];
    a1 += c * hr[(size_t)r * H_ + h1];
  }
  c_t[(size_t)b * H_ + h0] = a0;
  c_t[(size_t)b * H_ + h1] = a1;
}

extern "C" void kernel_launch(void* const* d_in, const int* in_sizes, int n_in,
                              void* d_out, int out_size, void* d_ws, size_t ws_size,
                              hipStream_t stream) {
  (void)in_sizes; (void)n_in; (void)out_size; (void)ws_size;
  const float* h_q    = (const float*)d_in[0];
  const float* q_mask = (const float*)d_in[1];
  const float* h_r    = (const float*)d_in[2];
  const float* r_mask = (const float*)d_in[3];
  const float* s_t    = (const float*)d_in[4];
  const float* qr_cov = (const float*)d_in[5];
  const float* U_w    = (const float*)d_in[6];
  const float* Wc_w   = (const float*)d_in[7];
  const float* Ws_w   = (const float*)d_in[8];
  const float* Wqr_w  = (const float*)d_in[9];
  const float* Wqr_b  = (const float*)d_in[10];
  const float* Vr_w   = (const float*)d_in[11];

  char* ws = (char*)d_ws;
  unsigned short* hq_bf  = (unsigned short*)(ws);             // 4,194,304 B
  unsigned short* hr_bf  = (unsigned short*)(ws + 4194304);   // 67,108,864 B
  unsigned short* Uw_bf  = (unsigned short*)(ws + 71303168);  // 524,288 B
  unsigned short* Wqr_bf = (unsigned short*)(ws + 71827456);  // 524,288 B
  float* alpha_q = (float*)(ws + 72351744);                   // 131,072 B
  float* alpha_r = (float*)(ws + 72482816);                   // 262,144 B
  float* wrow    = (float*)(ws + 72744960);                   // 278,528 B
  float* sfea    = (float*)(ws + 73023488);                   // 65,536 B
  float* logits  = (float*)(ws + 73089024);                   // 278,528 B
  float* coef    = (float*)(ws + 73367552);                   // 278,528 B

  float* out = (float*)d_out;
  float* c_t = out;
  float* a_out = out + B_ * H_;
  float* cov_out = a_out + B_ * L_;

  k_convert_bf16<<<2048, 256, 0, stream>>>(h_q, hq_bf, B_ * LQ_ * H_);
  k_convert_bf16<<<4096, 256, 0, stream>>>(h_r, hr_bf, B_ * K_ * LR_ * H_);
  k_convert_bf16<<<256, 256, 0, stream>>>(U_w, Uw_bf, H_ * H_);
  k_convert_bf16<<<256, 256, 0, stream>>>(Wqr_w, Wqr_bf, H_ * H_);

  k1_coatt<<<BK_, 256, 135680, stream>>>(hq_bf, hr_bf, Uw_bf, alpha_q, alpha_r);
  k2_wrow<<<(B_ * L_ + 255) / 256, 256, 0, stream>>>(alpha_q, alpha_r, q_mask, r_mask, wrow);
  k2_sfea<<<(B_ * H_ + 255) / 256, 256, 0, stream>>>(s_t, Ws_w, sfea);
  k3_logits<<<B_ * (L_ / 16), 256, 0, stream>>>(hq_bf, hr_bf, Wqr_bf, sfea, Wqr_b,
                                                Wc_w, Vr_w, wrow, qr_cov, logits);
  k4_softmax<<<B_, 256, 0, stream>>>(logits, q_mask, r_mask, qr_cov, wrow,
                                     a_out, cov_out, coef);
  k5_ct<<<B_, 256, 0, stream>>>(h_q, h_r, coef, c_t);
}
